// PointASNLSetAbstraction_84963043050009
// MI455X (gfx1250) — matrix operand order repaired and checked by reading, not compiled, not run on hardware
//
#include <hip/hip_runtime.h>
#include <hip/hip_bf16.h>
#include <math.h>

// ---------------------------------------------------------------------------
// PointASNL set abstraction, full forward pass for MI455X (gfx1250).
// Dense linears + nonlocal attention on v_wmma_f32_16x16x32_f16 (f16 operands,
// f32 accumulate). All GEMM operands are zero-padded to K%32==0 so the inner
// loop has no bounds logic at all. Irregular stages are scalar kernels.
// ---------------------------------------------------------------------------

typedef __attribute__((ext_vector_type(16))) _Float16 v16h;
typedef __attribute__((ext_vector_type(8)))  _Float16 v8h;
typedef __attribute__((ext_vector_type(8)))  float    v8f;

#define B_   8
#define N_   4096
#define S_   1024
#define NS_  32
#define AK_  8
#define CIN_ 64
#define CG_  67     // 3 + 64

// ---------------------------------------------------------------------------
// WMMA fragment loaders (per CDNA5 ISA §7.12.2 layouts, wave32)
// A tile in LDS: 16 rows x 32 halves, row-major (64B rows).
// lane<16: row=lane,    K = {0..7, 16..23}
// lane>=16: row=lane-16, K = {8..15, 24..31}
__device__ __forceinline__ v16h frag_a16(const _Float16* tile, int lane) {
  int r = lane & 15, hi = lane >> 4;
  const _Float16* p = tile + r * 32 + hi * 8;
  union { v16h v; v8h h[2]; } u;
  u.h[0] = *(const v8h*)(p);
  u.h[1] = *(const v8h*)(p + 16);
  return u.v;
}
// B tile in LDS: 16 columns x 32 K, column n stored contiguously at n*32.
// lane<16: col=lane, K = {0..7, 16..23} ; lane>=16: col=lane-16, K = {8..15, 24..31}
__device__ __forceinline__ v16h frag_b16(const _Float16* tile, int lane) {
  int n = lane & 15, hi = lane >> 4;
  const _Float16* p = tile + n * 32 + hi * 8;
  union { v16h v; v8h h[2]; } u;
  u.h[0] = *(const v8h*)(p);
  u.h[1] = *(const v8h*)(p + 16);
  return u.v;
}

// Stage 16 f32 -> 16 f16 into LDS (two 16B ds stores), fully unpredicated.
__device__ __forceinline__ void stage16_fast(_Float16* dst, const float* __restrict__ src) {
  v8h h0, h1;
  #pragma unroll
  for (int j = 0; j < 8; ++j) h0[j] = (_Float16)src[j];
  #pragma unroll
  for (int j = 0; j < 8; ++j) h1[j] = (_Float16)src[8 + j];
  *(v8h*)(dst)     = h0;
  *(v8h*)(dst + 8) = h1;
}

// Convert float4 pair -> v8h
__device__ __forceinline__ v8h pack8(float4 a, float4 b) {
  v8h h;
  h[0] = (_Float16)a.x; h[1] = (_Float16)a.y; h[2] = (_Float16)a.z; h[3] = (_Float16)a.w;
  h[4] = (_Float16)b.x; h[5] = (_Float16)b.y; h[6] = (_Float16)b.z; h[7] = (_Float16)b.w;
  return h;
}

// ---------------------------------------------------------------------------
// Generic WMMA GEMM:  C[M,N] = A[M,K] (row-major, lda) * W[N,K]^T + bias[N]
// Preconditions: M % 64 == 0, K % 32 == 0, A/W zero-padded in K (lda,ldw >= K).
// Block = 128 threads (4 waves). Wave w owns rows m0+16w; each wave computes
// TWO adjacent 16-wide N tiles sharing one A fragment (waves 0/1 stage them).
__global__ void wmma_gemm_bias(const float* __restrict__ A,
                               const float* __restrict__ W,
                               const float* __restrict__ bias,
                               float* __restrict__ C,
                               int M, int K, int N, int lda, int ldw) {
  __shared__ alignas(16) _Float16 lA[4][16 * 32];
  __shared__ alignas(16) _Float16 lB[2][16 * 32];
  int wave = threadIdx.x >> 5;
  int lane = threadIdx.x & 31;
  int m0 = blockIdx.x * 64 + wave * 16;
  int n0 = blockIdx.y * 32;
  v8f acc0 = {}, acc1 = {};
  int row  = lane & 15;
  int colh = (lane >> 4) * 16;  // this lane stages halves [colh, colh+16)
  const float* srcA = A + (size_t)(m0 + row) * lda + colh;
  int nrow = n0 + (wave & 1) * 16 + row;        // row of W staged by waves 0/1
  nrow = nrow < N ? nrow : N - 1;               // clamp (no predication)
  const float* srcW = W + (size_t)nrow * ldw + colh;
  for (int k0 = 0; k0 < K; k0 += 32) {
    stage16_fast(&lA[wave][row * 32 + colh], srcA + k0);
    if (k0 + 32 < K) __builtin_prefetch(srcA + k0 + 32, 0, 1);
    if (wave < 2) stage16_fast(&lB[wave][row * 32 + colh], srcW + k0);
    __syncthreads();
    v16h a  = frag_a16(lA[wave], lane);
    v16h b0 = frag_b16(lB[0], lane);
    v16h b1 = frag_b16(lB[1], lane);
    acc0 = __builtin_amdgcn_wmma_f32_16x16x32_f16(false, a, false, b0,
                                                  (short)0, acc0, false, false);
    acc1 = __builtin_amdgcn_wmma_f32_16x16x32_f16(false, a, false, b1,
                                                  (short)0, acc1, false, false);
    __syncthreads();
  }
  // store: C element (m = r + 8*(lane>>4), n = lane&15); M%64==0 so no m check
  int nA = n0 + (lane & 15);
  int nB = nA + 16;
  int mb = (lane >> 4) * 8;
  if (nA < N) {
    float bv = bias ? bias[nA] : 0.0f;
    #pragma unroll
    for (int r = 0; r < 8; ++r)
      C[(size_t)(m0 + mb + r) * N + nA] = acc0[r] + bv;
  }
  if (nB < N) {
    float bv = bias ? bias[nB] : 0.0f;
    #pragma unroll
    for (int r = 0; r < 8; ++r)
      C[(size_t)(m0 + mb + r) * N + nB] = acc1[r] + bv;
  }
}

// Repack weight (O x K, row-major) into zero-padded (O x Kp)
__global__ void pad_w_kernel(const float* __restrict__ src, float* __restrict__ dst,
                             int O, int K, int Kp) {
  int i = blockIdx.x * blockDim.x + threadIdx.x;
  if (i >= O * Kp) return;
  int o = i / Kp, k = i % Kp;
  dst[i] = (k < K) ? src[o * K + k] : 0.0f;
}

// ---------------------------------------------------------------------------
// BatchNorm: per-channel stats over all M rows, then affine (+ optional relu)
__global__ void bn_stats(const float* __restrict__ X, const float* __restrict__ g,
                         const float* __restrict__ be, float* __restrict__ scale,
                         float* __restrict__ shift, int M, int O) {
  __shared__ float ss[256], sq[256];
  int o = blockIdx.x, t = threadIdx.x;
  float s = 0.f, q = 0.f;
  for (int m = t; m < M; m += 256) {
    float v = X[(size_t)m * O + o];
    s += v; q += v * v;
  }
  ss[t] = s; sq[t] = q; __syncthreads();
  for (int st = 128; st > 0; st >>= 1) {
    if (t < st) { ss[t] += ss[t + st]; sq[t] += sq[t + st]; }
    __syncthreads();
  }
  if (t == 0) {
    float mu  = ss[0] / (float)M;
    float var = sq[0] / (float)M - mu * mu;
    float sc  = g[o] * rsqrtf(var + 1e-5f);
    scale[o] = sc;
    shift[o] = be[o] - mu * sc;
  }
}
__global__ void bn_apply(float* __restrict__ X, const float* __restrict__ scale,
                         const float* __restrict__ shift, size_t total, int O, int relu) {
  size_t i = (size_t)blockIdx.x * blockDim.x + threadIdx.x;
  if (i >= total) return;
  int o = (int)(i % O);
  float v = X[i] * scale[o] + shift[o];
  X[i] = relu ? fmaxf(v, 0.0f) : v;
}

// ---------------------------------------------------------------------------
// Farthest point sampling: 1 block per batch, 512 threads, 1024 sequential steps
__global__ void fps_kernel(const float* __restrict__ xyz, int* __restrict__ fidx,
                           float* __restrict__ nx) {
  __shared__ float sv[512]; __shared__ int si[512];
  __shared__ float cc[3];   __shared__ int sfar;
  int b = blockIdx.x, t = threadIdx.x;
  const float* base = xyz + (size_t)b * N_ * 3;
  float dist[8];
  #pragma unroll
  for (int j = 0; j < 8; ++j) dist[j] = 1e10f;
  if (t == 0) sfar = 0;
  __syncthreads();
  for (int it = 0; it < S_; ++it) {
    if (t == 0) {
      int far = sfar;
      fidx[b * S_ + it] = far;
      cc[0] = base[far * 3 + 0]; cc[1] = base[far * 3 + 1]; cc[2] = base[far * 3 + 2];
      nx[(b * S_ + it) * 3 + 0] = cc[0];
      nx[(b * S_ + it) * 3 + 1] = cc[1];
      nx[(b * S_ + it) * 3 + 2] = cc[2];
    }
    __syncthreads();
    float cx = cc[0], cy = cc[1], cz = cc[2];
    float bv = -1.0f; int bi = 0;
    #pragma unroll
    for (int j = 0; j < 8; ++j) {
      int n = j * 512 + t;
      float dx = base[n * 3 + 0] - cx, dy = base[n * 3 + 1] - cy, dz = base[n * 3 + 2] - cz;
      float d = dx * dx + dy * dy + dz * dz;
      float dn = fminf(dist[j], d); dist[j] = dn;
      if (dn > bv) { bv = dn; bi = n; }
    }
    sv[t] = bv; si[t] = bi; __syncthreads();
    for (int st = 256; st > 0; st >>= 1) {
      if (t < st) {
        float ov = sv[t + st]; int oi = si[t + st];
        if (ov > sv[t] || (ov == sv[t] && oi < si[t])) { sv[t] = ov; si[t] = oi; }
      }
      __syncthreads();
    }
    if (t == 0) sfar = si[0];
    __syncthreads();
  }
}

// KNN: one thread per query, sorted insertion top-32 (ascending d2, stable ties)
__global__ void knn_kernel(const float* __restrict__ xyz, const float* __restrict__ q,
                           int* __restrict__ idx) {
  int gid = blockIdx.x * blockDim.x + threadIdx.x;
  if (gid >= B_ * S_) return;
  int b = gid / S_;
  const float* base = xyz + (size_t)b * N_ * 3;
  float qx = q[gid * 3 + 0], qy = q[gid * 3 + 1], qz = q[gid * 3 + 2];
  float bd[NS_]; int bi[NS_];
  #pragma unroll
  for (int i = 0; i < NS_; ++i) { bd[i] = 3.4e38f; bi[i] = 0; }
  for (int n = 0; n < N_; ++n) {
    float dx = base[n * 3 + 0] - qx, dy = base[n * 3 + 1] - qy, dz = base[n * 3 + 2] - qz;
    float d2 = dx * dx + dy * dy + dz * dz;
    if (d2 < bd[NS_ - 1]) {
      int j = NS_ - 1;
      while (j > 0 && bd[j - 1] > d2) { bd[j] = bd[j - 1]; bi[j] = bi[j - 1]; --j; }
      bd[j] = d2; bi[j] = n;
    }
  }
  for (int i = 0; i < NS_; ++i) idx[(size_t)gid * NS_ + i] = bi[i];
}

// Gather grouped xyz / feature (feature rows are 64 floats = 16B aligned)
__global__ void gather_kernel(const float* __restrict__ xyz, const float* __restrict__ feat,
                              const int* __restrict__ idx, float* __restrict__ gxyz,
                              float* __restrict__ gfeat) {
  size_t g = (size_t)blockIdx.x * blockDim.x + threadIdx.x;
  if (g >= (size_t)B_ * S_ * NS_) return;
  int b = (int)(g / ((size_t)S_ * NS_));
  int id = idx[g];
  const float* px = xyz + ((size_t)b * N_ + id) * 3;
  gxyz[g * 3 + 0] = px[0]; gxyz[g * 3 + 1] = px[1]; gxyz[g * 3 + 2] = px[2];
  const float4* pf = (const float4*)(feat + ((size_t)b * N_ + id) * CIN_);
  float4* o = (float4*)(gfeat + g * CIN_);
  #pragma unroll
  for (int c = 0; c < CIN_ / 4; ++c) o[c] = pf[c];
}

// Adaptive-sampling input: x70 rows (lda 96) = [nxyz(3), gxyz(3), gfeat(64), 0...]
__global__ void build_x70_kernel(const float* __restrict__ gxyz,
                                 const float* __restrict__ gfeat,
                                 float* __restrict__ x70) {
  int r = blockIdx.x * blockDim.x + threadIdx.x;   // (b*S+s)*8 + k
  if (r >= B_ * S_ * AK_) return;
  int bs = r / AK_, k = r % AK_;
  size_t gk = ((size_t)bs * NS_ + k) * 3, g0 = (size_t)bs * NS_ * 3;
  float* o = x70 + (size_t)r * 96;
  for (int d = 0; d < 3; ++d) { o[d] = gxyz[gk + d] - gxyz[g0 + d]; o[3 + d] = gxyz[gk + d]; }
  const float* f = gfeat + ((size_t)bs * NS_ + k) * CIN_;
  for (int c = 0; c < CIN_; ++c) o[6 + c] = f[c];
  for (int c = 70; c < 96; ++c) o[c] = 0.0f;
}

// Per-point 8x8 attention: g = softmax(tnp . f1^T / sqrt(33)) @ feat
// writes g33 rows zero-padded to lda 64
__global__ void as_attn_kernel(const float* __restrict__ tnp33,
                               const float* __restrict__ tf66,
                               float* __restrict__ g33) {
  int r = blockIdx.x * blockDim.x + threadIdx.x;   // row (b*S+s)*8 + k
  if (r >= B_ * S_ * AK_) return;
  int base = (r / AK_) * AK_;
  const float scale = 0.17407766f;                 // 1/sqrt(33)
  float sc[AK_];
  float mx = -3.0e38f;
  for (int j = 0; j < AK_; ++j) {
    const float* a = tnp33 + (size_t)r * 33;
    const float* f = tf66 + (size_t)(base + j) * 66;
    float s = 0.f;
    for (int c = 0; c < 33; ++c) s += a[c] * f[c];
    sc[j] = s * scale;
    mx = fmaxf(mx, sc[j]);
  }
  float sum = 0.f;
  for (int j = 0; j < AK_; ++j) { sc[j] = __expf(sc[j] - mx); sum += sc[j]; }
  float inv = 1.0f / sum;
  float* o = g33 + (size_t)r * 64;
  for (int c = 0; c < 33; ++c) {
    float acc = 0.f;
    for (int j = 0; j < AK_; ++j) acc += sc[j] * tf66[(size_t)(base + j) * 66 + 33 + c];
    o[c] = acc * inv;
  }
  for (int c = 33; c < 64; ++c) o[c] = 0.0f;
}

// softmax over axis=1 (the S axis) of tensor (B, S, 8, 68)
__global__ void softmax_axis1_kernel(float* __restrict__ X) {
  int id = blockIdx.x * blockDim.x + threadIdx.x;  // b*(8*68) + k*68 + c
  if (id >= B_ * AK_ * 68) return;
  int b = id / (AK_ * 68), rem = id % (AK_ * 68);
  size_t base = ((size_t)b * S_) * (AK_ * 68) + rem;
  size_t stride = AK_ * 68;
  float mx = -3.0e38f;
  for (int s = 0; s < S_; ++s) mx = fmaxf(mx, X[base + s * stride]);
  float sum = 0.f;
  for (int s = 0; s < S_; ++s) sum += __expf(X[base + s * stride] - mx);
  float inv = 1.0f / sum;
  for (int s = 0; s < S_; ++s) {
    size_t a = base + s * stride;
    X[a] = __expf(X[a] - mx) * inv;
  }
}

// new_xyz = sum_k sx*sw[...,0] ; new_feat (lda 96, zero-padded) = sum_k sf*sw[...,1:]
__global__ void as_combine_kernel(const float* __restrict__ gxyz,
                                  const float* __restrict__ gfeat,
                                  const float* __restrict__ sw68,
                                  float* __restrict__ nxout,
                                  float* __restrict__ nfeat) {
  int bs = blockIdx.x * blockDim.x + threadIdx.x;
  if (bs >= B_ * S_) return;
  float ax = 0.f, ay = 0.f, az = 0.f;
  for (int k = 0; k < AK_; ++k) {
    const float* w = sw68 + ((size_t)bs * AK_ + k) * 68;
    const float* x = gxyz + ((size_t)bs * NS_ + k) * 3;
    ax += x[0] * w[0]; ay += x[1] * w[0]; az += x[2] * w[0];
  }
  nxout[(size_t)bs * 3 + 0] = ax;
  nxout[(size_t)bs * 3 + 1] = ay;
  nxout[(size_t)bs * 3 + 2] = az;
  float* o = nfeat + (size_t)bs * 96;
  for (int c = 0; c < CG_; ++c) {
    float acc = 0.f;
    for (int k = 0; k < AK_; ++k) {
      const float* w = sw68 + ((size_t)bs * AK_ + k) * 68;
      float sf = (c < 3) ? gxyz[((size_t)bs * NS_ + k) * 3 + c]
                         : gfeat[((size_t)bs * NS_ + k) * CIN_ + (c - 3)];
      acc += sf * w[1 + c];
    }
    o[c] = acc;
  }
  for (int c = CG_; c < 96; ++c) o[c] = 0.0f;
}

// new_point (lda 96, zero-padded) = [gxyz - new_xyz (3), gfeat (64), 0...]
// also writes gxn3 (lda 32): [gxyz - new_xyz (3), 0...] for the weight net
__global__ void build_np_kernel(const float* __restrict__ gxyz,
                                const float* __restrict__ gfeat,
                                const float* __restrict__ nxout,
                                float* __restrict__ np,
                                float* __restrict__ gx3) {
  size_t r = (size_t)blockIdx.x * blockDim.x + threadIdx.x;  // (b*S+s)*NS + k
  if (r >= (size_t)B_ * S_ * NS_) return;
  size_t bs = r / NS_;
  float* o = np + r * 96;
  float* o3 = gx3 + r * 32;
  for (int d = 0; d < 3; ++d) {
    float v = gxyz[r * 3 + d] - nxout[bs * 3 + d];
    o[d] = v; o3[d] = v;
  }
  for (int d = 3; d < 32; ++d) o3[d] = 0.0f;
  const float* f = gfeat + r * CIN_;
  for (int c = 0; c < CIN_; ++c) o[3 + c] = f[c];
  for (int c = CG_; c < 96; ++c) o[c] = 0.0f;
}

// ---------------------------------------------------------------------------
// Fused nonlocal attention (flash style): one wave per 16-query tile.
//   Q = tnp2[b, s, 0..31]; K = tfm[b, n, 0..31]; V = tfm[b, n, 32..63]
__global__ void nl_attention_kernel(const float* __restrict__ Qm,
                                    const float* __restrict__ Fm,
                                    float* __restrict__ Out) {
  __shared__ alignas(16) _Float16 qa[16 * 32];
  __shared__ alignas(16) _Float16 kb[16 * 32];
  __shared__ alignas(16) _Float16 pa[16 * 32];
  __shared__ alignas(16) _Float16 vb[2][16 * 32];
  __shared__ float sS[16 * 16];
  __shared__ float cf_s[16];
  __shared__ float l_s[16];
  int lane = threadIdx.x;
  int b = blockIdx.y, q0 = blockIdx.x * 16;
  // zero K-pad regions once (j >= 16 of pa and vb are never re-written)
  for (int i = lane; i < 16 * 32; i += 32) pa[i] = (_Float16)0.0f;
  for (int i = lane; i < 2 * 16 * 32; i += 32) ((_Float16*)vb)[i] = (_Float16)0.0f;
  {  // stage Q tile once: row r, channels c0..c0+15 (vectorized)
    int r = lane & 15, c0 = (lane >> 4) * 16;
    const float* src = Qm + ((size_t)b * S_ + q0 + r) * 32 + c0;
    float4 f0 = *(const float4*)(src);
    float4 f1 = *(const float4*)(src + 4);
    float4 f2 = *(const float4*)(src + 8);
    float4 f3 = *(const float4*)(src + 12);
    *(v8h*)(&qa[r * 32 + c0])     = pack8(f0, f1);
    *(v8h*)(&qa[r * 32 + c0 + 8]) = pack8(f2, f3);
  }
  float m_i = -3.0e38f, l_i = 0.0f;  // valid in lanes 0..15 (row = lane)
  v8f o0 = {}, o1 = {};
  const float scale = 0.17677670f;   // 1/sqrt(32)
  for (int n0 = 0; n0 < N_; n0 += 16) {
    {  // stage K tile: B col n = key n0+n, channels 0..31 (vectorized)
      int n = lane & 15, c0 = (lane >> 4) * 16;
      const float* src = Fm + ((size_t)b * N_ + n0 + n) * 64 + c0;
      float4 f0 = *(const float4*)(src);
      float4 f1 = *(const float4*)(src + 4);
      float4 f2 = *(const float4*)(src + 8);
      float4 f3 = *(const float4*)(src + 12);
      *(v8h*)(&kb[n * 32 + c0])     = pack8(f0, f1);
      *(v8h*)(&kb[n * 32 + c0 + 8]) = pack8(f2, f3);
      if (n0 + 16 < N_) __builtin_prefetch(src + 16 * 64, 0, 1);
    }
    {  // stage V tile transposed: lane reads row (n0+r), 16 contiguous channels,
       // scatters into B layout vb[half][c*32 + r]
      int r = lane & 15, half = lane >> 4;
      const float* src = Fm + ((size_t)b * N_ + n0 + r) * 64 + 32 + half * 16;
      float4 f0 = *(const float4*)(src);
      float4 f1 = *(const float4*)(src + 4);
      float4 f2 = *(const float4*)(src + 8);
      float4 f3 = *(const float4*)(src + 12);
      _Float16* vd = vb[half];
      vd[0 * 32 + r]  = (_Float16)f0.x; vd[1 * 32 + r]  = (_Float16)f0.y;
      vd[2 * 32 + r]  = (_Float16)f0.z; vd[3 * 32 + r]  = (_Float16)f0.w;
      vd[4 * 32 + r]  = (_Float16)f1.x; vd[5 * 32 + r]  = (_Float16)f1.y;
      vd[6 * 32 + r]  = (_Float16)f1.z; vd[7 * 32 + r]  = (_Float16)f1.w;
      vd[8 * 32 + r]  = (_Float16)f2.x; vd[9 * 32 + r]  = (_Float16)f2.y;
      vd[10 * 32 + r] = (_Float16)f2.z; vd[11 * 32 + r] = (_Float16)f2.w;
      vd[12 * 32 + r] = (_Float16)f3.x; vd[13 * 32 + r] = (_Float16)f3.y;
      vd[14 * 32 + r] = (_Float16)f3.z; vd[15 * 32 + r] = (_Float16)f3.w;
    }
    __syncthreads();
    v16h aQ = frag_a16(qa, lane);
    v16h bK = frag_b16(kb, lane);
    v8f s = {};
    s = __builtin_amdgcn_wmma_f32_16x16x32_f16(false, aQ, false, bK,
                                               (short)0, s, false, false);
    {  // spill S tile to LDS for row-wise online softmax
      int n = lane & 15, mb = (lane >> 4) * 8;
      #pragma unroll
      for (int r = 0; r < 8; ++r) sS[(mb + r) * 16 + n] = s[r] * scale;
    }
    __syncthreads();
    if (lane < 16) {  // online softmax per row
      int m = lane;
      float rmax = -3.0e38f;
      for (int n = 0; n < 16; ++n) rmax = fmaxf(rmax, sS[m * 16 + n]);
      float nm = fmaxf(m_i, rmax);
      float cf = __expf(m_i - nm);
      float ps = 0.f;
      v8h p0, p1;
      #pragma unroll
      for (int n = 0; n < 8; ++n) {
        float p = __expf(sS[m * 16 + n] - nm);
        ps += p; p0[n] = (_Float16)p;
      }
      #pragma unroll
      for (int n = 0; n < 8; ++n) {
        float p = __expf(sS[m * 16 + 8 + n] - nm);
        ps += p; p1[n] = (_Float16)p;
      }
      *(v8h*)(&pa[m * 32])     = p0;
      *(v8h*)(&pa[m * 32 + 8]) = p1;
      l_i = l_i * cf + ps; m_i = nm; cf_s[m] = cf;
    }
    __syncthreads();
    {  // rescale accumulators (rows of this lane are m = 8*(lane>>4)+r)
      int hi = lane >> 4;
      #pragma unroll
      for (int r = 0; r < 8; ++r) {
        float cf = cf_s[hi * 8 + r];
        o0[r] *= cf; o1[r] *= cf;
      }
    }
    v16h aP  = frag_a16(pa, lane);
    v16h bV0 = frag_b16(vb[0], lane);
    v16h bV1 = frag_b16(vb[1], lane);
    o0 = __builtin_amdgcn_wmma_f32_16x16x32_f16(false, aP, false, bV0,
                                                (short)0, o0, false, false);
    o1 = __builtin_amdgcn_wmma_f32_16x16x32_f16(false, aP, false, bV1,
                                                (short)0, o1, false, false);
    __syncthreads();
  }
  if (lane < 16) l_s[lane] = l_i;
  __syncthreads();
  {
    int n = lane & 15, hi = lane >> 4;
    #pragma unroll
    for (int r = 0; r < 8; ++r) {
      int m = hi * 8 + r;
      float inv = 1.0f / l_s[m];
      Out[((size_t)b * S_ + q0 + m) * 32 + n]      = o0[r] * inv;
      Out[((size_t)b * S_ + q0 + m) * 32 + 16 + n] = o1[r] * inv;
    }
  }
}

// skip = max over k of new_point (covers pad columns too: np pad is 0 -> max 0)
__global__ void skip_max_kernel(const float* __restrict__ np, float* __restrict__ sk) {
  size_t i = (size_t)blockIdx.x * blockDim.x + threadIdx.x;  // (b*S+s)*96 + c
  if (i >= (size_t)B_ * S_ * 96) return;
  size_t bs = i / 96; int c = (int)(i % 96);
  float mx = -3.0e38f;
  for (int k = 0; k < NS_; ++k)
    mx = fmaxf(mx, np[((size_t)bs * NS_ + k) * 96 + c]);
  sk[i] = mx;
}

// agg[b,s,c*32+w] = sum_k x[b,s,k,c] * w[b,s,k,w]
__global__ void agg_kernel(const float* __restrict__ X, const float* __restrict__ W,
                           float* __restrict__ Agg) {
  __shared__ float sx[32 * 64];
  __shared__ float sw[32 * 32];
  int bs = blockIdx.x;
  {
    const float4* xs = (const float4*)(X + (size_t)bs * 2048);
    const float4* wsrc = (const float4*)(W + (size_t)bs * 1024);
    float4* xd = (float4*)sx;
    float4* wd = (float4*)sw;
    for (int i = threadIdx.x; i < 512; i += 256) xd[i] = xs[i];
    for (int i = threadIdx.x; i < 256; i += 256) wd[i] = wsrc[i];
  }
  __syncthreads();
  for (int o = threadIdx.x; o < 2048; o += 256) {
    int c = o >> 5, w = o & 31;
    float acc = 0.f;
    #pragma unroll
    for (int k = 0; k < 32; ++k) acc += sx[k * 64 + c] * sw[k * 32 + w];
    Agg[(size_t)bs * 2048 + o] = acc;
  }
}

// out = relu(relu(bn(final)) + skip + nl)
__global__ void final_combine_kernel(const float* __restrict__ fin,
                                     const float* __restrict__ sk,
                                     const float* __restrict__ nl,
                                     float* __restrict__ out) {
  size_t i = (size_t)blockIdx.x * blockDim.x + threadIdx.x;
  if (i >= (size_t)B_ * S_ * 128) return;
  out[i] = fmaxf(fin[i] + sk[i] + nl[i], 0.0f);
}

// ---------------------------------------------------------------------------
struct Lin { const float *b, *be, *g, *w; };
static Lin lin_at(void* const* d, int i) {
  Lin l = { (const float*)d[i], (const float*)d[i + 1],
            (const float*)d[i + 2], (const float*)d[i + 3] };
  return l;
}
static void gemm(hipStream_t st, const float* A, const float* W, const float* bias,
                 float* C, int M, int K, int N, int lda, int ldw) {
  dim3 g((M + 63) / 64, (N + 31) / 32);
  wmma_gemm_bias<<<g, 128, 0, st>>>(A, W, bias, C, M, K, N, lda, ldw);
}
static void bn(hipStream_t st, float* X, const float* g_, const float* be,
               float* sc, float* sh, int M, int O, int relu) {
  bn_stats<<<O, 256, 0, st>>>(X, g_, be, sc, sh, M, O);
  size_t tot = (size_t)M * O;
  bn_apply<<<(unsigned)((tot + 255) / 256), 256, 0, st>>>(X, sc, sh, tot, O, relu);
}
static void padw(hipStream_t st, const float* src, float* dst, int O, int K, int Kp) {
  pad_w_kernel<<<(O * Kp + 255) / 256, 256, 0, st>>>(src, dst, O, K, Kp);
}

extern "C" void kernel_launch(void* const* d_in, const int* in_sizes, int n_in,
                              void* d_out, int out_size, void* d_ws, size_t ws_size,
                              hipStream_t stream) {
  (void)in_sizes; (void)n_in; (void)out_size; (void)ws_size;
  const float* xyz     = (const float*)d_in[0];
  const float* feature = (const float*)d_in[1];
  // JAX sorted-dict tree-flatten order; each leaf dict flattens as b, be, g, w
  Lin as_mlp0 = lin_at(d_in, 2);
  Lin as_mlp1 = lin_at(d_in, 6);
  Lin as_tf   = lin_at(d_in, 10);
  Lin as_tnp  = lin_at(d_in, 14);
  Lin fin     = lin_at(d_in, 18);
  Lin mlp0    = lin_at(d_in, 22);
  Lin mlp1    = lin_at(d_in, 26);
  Lin nl_nnp  = lin_at(d_in, 30);
  Lin nl_tf   = lin_at(d_in, 34);
  Lin nl_tnp  = lin_at(d_in, 38);
  Lin skipL   = lin_at(d_in, 42);
  Lin wn0     = lin_at(d_in, 46);

  float* ws = (float*)d_ws;
  size_t off = 0;
  auto A = [&](size_t n) { float* p = ws + off; off += n; return p; };

  int*   fidx  = (int*)A((size_t)B_ * S_);
  float* nxfps = A((size_t)B_ * S_ * 3);
  int*   knn   = (int*)A((size_t)B_ * S_ * NS_);
  float* gxyz  = A((size_t)B_ * S_ * NS_ * 3);
  float* regG  = A((size_t)B_ * S_ * NS_ * CIN_);   // gfeat -> xa -> wn (reuse)
  float* regR  = A((size_t)23527424);               // adaptive interm. -> gxn3 -> agg
  float* nfeat = A((size_t)B_ * S_ * 96);
  float* npnt  = A((size_t)B_ * S_ * NS_ * 96);
  float* tfm   = A((size_t)B_ * N_ * 64);
  float* tnp2  = A((size_t)B_ * S_ * 32);
  float* nl0   = A((size_t)B_ * S_ * 32);
  float* nl128 = A((size_t)B_ * S_ * 128);
  float* sk96  = A((size_t)B_ * S_ * 96);
  float* sko   = A((size_t)B_ * S_ * 128);
  float* xb    = A((size_t)B_ * S_ * NS_ * 64);
  float* finb  = A((size_t)B_ * S_ * 128);
  float* bnsc  = A(256);
  float* bnsh  = A(256);
  // zero-padded weight copies (K padded to 32)
  float* wp_astf   = A(66 * 96);
  float* wp_astnp  = A(33 * 96);
  float* wp_asmlp0 = A(32 * 64);
  float* wp_mlp0   = A(64 * 96);
  float* wp_nltnp  = A(32 * 96);
  float* wp_skip   = A(128 * 96);
  float* wp_wn     = A(32 * 32);
  // region R sublayout (all dead before gxn3/agg reuse the base)
  float* x70   = regR;                              // 65536 x 96
  float* tf66  = x70 + (size_t)65536 * 96;          // 65536 x 66
  float* tnp33 = tf66 + (size_t)65536 * 66;         // 65536 x 33
  float* g33   = tnp33 + (size_t)65536 * 33;        // 65536 x 64 (padded)
  float* m32   = g33 + (size_t)65536 * 64;          // 65536 x 32
  float* sw68  = m32 + (size_t)65536 * 32;          // 65536 x 68
  float* gfeat = regG;
  float* xa    = regG;   // reuse after gfeat dead
  float* wnb   = regG;   // reuse after xa dead
  float* gxn3  = regR;   // reuse after adaptive stage dead (262144 x 32)
  float* agg   = regR;   // reuse after wn GEMM consumed gxn3
  float* nxout = (float*)d_out;                       // (B, 1024, 3)
  float* outf  = (float*)d_out + (size_t)B_ * S_ * 3; // (B, 1024, 128)

  // 0) repack ragged-K weights into zero-padded copies
  padw(stream, as_tf.w, wp_astf, 66, 70, 96);
  padw(stream, as_tnp.w, wp_astnp, 33, 70, 96);
  padw(stream, as_mlp0.w, wp_asmlp0, 32, 33, 64);
  padw(stream, mlp0.w, wp_mlp0, 64, CG_, 96);
  padw(stream, nl_tnp.w, wp_nltnp, 32, CG_, 96);
  padw(stream, skipL.w, wp_skip, 128, CG_, 96);
  padw(stream, wn0.w, wp_wn, 32, 3, 32);

  // 1) sampling / grouping
  fps_kernel<<<B_, 512, 0, stream>>>(xyz, fidx, nxfps);
  knn_kernel<<<(B_ * S_ + 63) / 64, 64, 0, stream>>>(xyz, nxfps, knn);
  {
    size_t t = (size_t)B_ * S_ * NS_;
    gather_kernel<<<(unsigned)((t + 255) / 256), 256, 0, stream>>>(xyz, feature, knn, gxyz, gfeat);
  }
  // 2) adaptive sampling
  build_x70_kernel<<<(B_ * S_ * AK_ + 255) / 256, 256, 0, stream>>>(gxyz, gfeat, x70);
  gemm(stream, x70, wp_astf, as_tf.b, tf66, 65536, 96, 66, 96, 96);
  bn(stream, tf66, as_tf.g, as_tf.be, bnsc, bnsh, 65536, 66, 0);
  gemm(stream, x70, wp_astnp, as_tnp.b, tnp33, 65536, 96, 33, 96, 96);
  bn(stream, tnp33, as_tnp.g, as_tnp.be, bnsc, bnsh, 65536, 33, 0);
  as_attn_kernel<<<(B_ * S_ * AK_ + 255) / 256, 256, 0, stream>>>(tnp33, tf66, g33);
  gemm(stream, g33, wp_asmlp0, as_mlp0.b, m32, 65536, 64, 32, 64, 64);
  bn(stream, m32, as_mlp0.g, as_mlp0.be, bnsc, bnsh, 65536, 32, 1);
  gemm(stream, m32, as_mlp1.w, as_mlp1.b, sw68, 65536, 32, 68, 32, 32);
  bn(stream, sw68, as_mlp1.g, as_mlp1.be, bnsc, bnsh, 65536, 68, 0);
  softmax_axis1_kernel<<<(B_ * AK_ * 68 + 255) / 256, 256, 0, stream>>>(sw68);
  as_combine_kernel<<<(B_ * S_ + 255) / 256, 256, 0, stream>>>(gxyz, gfeat, sw68, nxout, nfeat);
  {
    size_t t = (size_t)B_ * S_ * NS_;
    build_np_kernel<<<(unsigned)((t + 255) / 256), 256, 0, stream>>>(gxyz, gfeat, nxout, npnt, gxn3);
  }
  // 3) nonlocal cell (flash attention over 4096 keys)
  gemm(stream, feature, nl_tf.w, nl_tf.b, tfm, B_ * N_, 64, 64, 64, 64);
  bn(stream, tfm, nl_tf.g, nl_tf.be, bnsc, bnsh, B_ * N_, 64, 0);
  gemm(stream, nfeat, wp_nltnp, nl_tnp.b, tnp2, B_ * S_, 96, 32, 96, 96);
  bn(stream, tnp2, nl_tnp.g, nl_tnp.be, bnsc, bnsh, B_ * S_, 32, 0);
  {
    dim3 g(S_ / 16, B_);
    nl_attention_kernel<<<g, 32, 0, stream>>>(tnp2, tfm, nl0);
  }
  gemm(stream, nl0, nl_nnp.w, nl_nnp.b, nl128, B_ * S_, 32, 128, 32, 32);
  bn(stream, nl128, nl_nnp.g, nl_nnp.be, bnsc, bnsh, B_ * S_, 128, 0);
  // 4) skip branch
  {
    size_t t = (size_t)B_ * S_ * 96;
    skip_max_kernel<<<(unsigned)((t + 255) / 256), 256, 0, stream>>>(npnt, sk96);
  }
  gemm(stream, sk96, wp_skip, skipL.b, sko, B_ * S_, 96, 128, 96, 96);
  bn(stream, sko, skipL.g, skipL.be, bnsc, bnsh, B_ * S_, 128, 1);
  // 5) point MLP chain + weight net + aggregation
  gemm(stream, npnt, wp_mlp0, mlp0.b, xa, B_ * S_ * NS_, 96, 64, 96, 96);
  bn(stream, xa, mlp0.g, mlp0.be, bnsc, bnsh, B_ * S_ * NS_, 64, 1);
  gemm(stream, xa, mlp1.w, mlp1.b, xb, B_ * S_ * NS_, 64, 64, 64, 64);
  bn(stream, xb, mlp1.g, mlp1.be, bnsc, bnsh, B_ * S_ * NS_, 64, 1);
  gemm(stream, gxn3, wp_wn, wn0.b, wnb, B_ * S_ * NS_, 32, 32, 32, 32);
  bn(stream, wnb, wn0.g, wn0.be, bnsc, bnsh, B_ * S_ * NS_, 32, 1);
  agg_kernel<<<B_ * S_, 256, 0, stream>>>(xb, wnb, agg);
  // 6) final projection + combine
  gemm(stream, agg, fin.w, fin.b, finb, B_ * S_, 2048, 128, 2048, 2048);
  bn(stream, finb, fin.g, fin.be, bnsc, bnsh, B_ * S_, 128, 1);
  {
    size_t t = (size_t)B_ * S_ * 128;
    final_combine_kernel<<<(unsigned)((t + 255) / 256), 256, 0, stream>>>(finb, sko, nl128, outf);
  }
}
